// SimScore_24850680774756
// MI455X (gfx1250) — matrix operand order repaired and checked by reading, not compiled, not run on hardware
//
#include <hip/hip_runtime.h>
#include <hip/hip_bf16.h>

typedef __bf16 bf16_t;
typedef __attribute__((ext_vector_type(16))) __bf16 v16bf;
typedef __attribute__((ext_vector_type(8)))  __bf16 v8bf;
typedef __attribute__((ext_vector_type(8)))  float  v8f;
typedef __attribute__((ext_vector_type(4)))  float  v4f;

#define N_B   64
#define S_SEQ 1024
#define E_DIM 512
#define H_DIM 512
#define BM    64
#define TEMP_INV (1.0f / 13.544f)

// ---- LDS layout (dynamic):
//   logits f32[64][LSTRIDE]  (exp'd in place; doubles as p in f32)
//   rowsum f32[64]           (1/sum(exp) per row)
//   hbuf   f32[64][HSTRIDE]  aliases the logits region AFTER the fc1 GEMM
#define LSTRIDE 1036                        // mult of 4 (16B rows) ; 1036%64=12 spreads banks
#define LOGITS_BYTES (BM * LSTRIDE * 4)     // 265216
#define ROWSUM_OFF   LOGITS_BYTES
#define SMEM_BYTES   (ROWSUM_OFF + BM * 4)  // 265472 < 320KB WGP LDS
#define HSTRIDE 517                         // 64*517*4 = 132352 <= logits region

__device__ __forceinline__ v16bf load16(const bf16_t* p0, const bf16_t* p1) {
    v8bf lo = *reinterpret_cast<const v8bf*>(p0);
    v8bf hi = *reinterpret_cast<const v8bf*>(p1);
    v16bf r;
#pragma unroll
    for (int i = 0; i < 8; ++i) { r[i] = lo[i]; r[i + 8] = hi[i]; }
    return r;
}

// f32 (LDS) -> bf16 A-fragment, converted in registers (v_cvt_pk_bf16_f32)
__device__ __forceinline__ v16bf cvt16(const float* p0, const float* p1) {
    v4f a0 = *reinterpret_cast<const v4f*>(p0);
    v4f a1 = *reinterpret_cast<const v4f*>(p0 + 4);
    v4f b0 = *reinterpret_cast<const v4f*>(p1);
    v4f b1 = *reinterpret_cast<const v4f*>(p1 + 4);
    v16bf r;
#pragma unroll
    for (int i = 0; i < 4; ++i) {
        r[i]      = (bf16_t)a0[i];
        r[i + 4]  = (bf16_t)a1[i];
        r[i + 8]  = (bf16_t)b0[i];
        r[i + 12] = (bf16_t)b1[i];
    }
    return r;
}

// ---- pre-pass: x -> bf16 + row squared-norms (f32) -------------------------
__global__ void conv_x_kernel(const float* __restrict__ x,
                              bf16_t* __restrict__ xbf,
                              float* __restrict__ sq) {
    const int row = blockIdx.x;            // one (b,s) row of 512
    const int t   = threadIdx.x;           // 128 threads * 4 elems
    const float4 v4 = reinterpret_cast<const float4*>(x + (size_t)row * E_DIM)[t];
    bf16_t* xo = xbf + (size_t)row * E_DIM + t * 4;
    xo[0] = (bf16_t)v4.x; xo[1] = (bf16_t)v4.y;
    xo[2] = (bf16_t)v4.z; xo[3] = (bf16_t)v4.w;
    float s = v4.x * v4.x + v4.y * v4.y + v4.z * v4.z + v4.w * v4.w;
#pragma unroll
    for (int d = 1; d < 32; d <<= 1) s += __shfl_xor(s, d, 32);
    __shared__ float wsum[4];
    if ((t & 31) == 0) wsum[t >> 5] = s;
    __syncthreads();
    if (t == 0) sq[row] = wsum[0] + wsum[1] + wsum[2] + wsum[3];
}

__global__ void conv_w1_kernel(const float* __restrict__ w1,
                               bf16_t* __restrict__ w1bf, int n) {
    const int i = blockIdx.x * blockDim.x + threadIdx.x;
    if (i < n) w1bf[i] = (bf16_t)w1[i];
}

// ---- fused: scores -> softmax -> fc1 -> LN -> relu -> fc2 -> relu ----------
__global__ __launch_bounds__(256, 1) void fused_kernel(
    const bf16_t* __restrict__ xbf, const float* __restrict__ sq,
    const bf16_t* __restrict__ w1bf, const float* __restrict__ b1,
    const float* __restrict__ g1, const float* __restrict__ beta1,
    const float* __restrict__ w2, const float* __restrict__ b2,
    float* __restrict__ out) {
    extern __shared__ char smem[];
    float* logits = reinterpret_cast<float*>(smem);
    float* rowsum = reinterpret_cast<float*>(smem + ROWSUM_OFF);
    float* hbuf   = reinterpret_cast<float*>(smem);   // reuses logits region late

    const int b    = blockIdx.x >> 4;      // batch
    const int rb   = blockIdx.x & 15;      // 64-row block within S
    const int tid  = threadIdx.x;
    const int lane = tid & 31;
    const int wave = tid >> 5;             // 8 waves
    const int rt   = wave & 3;             // 16-row tile within block (4 tiles)
    const int ch   = wave >> 2;            // key-interleave / col-half (2)
    const int ln   = lane & 15;
    const int half = lane >> 4;

    const int    rBase = rb * BM + rt * 16;
    const size_t xrow0 = (size_t)(b * S_SEQ + rBase);

    const v8f vzero = {0.f, 0.f, 0.f, 0.f, 0.f, 0.f, 0.f, 0.f};

    // ================= Phase 1: scaled neg-distance logits ==================
    float sqr[8];
#pragma unroll
    for (int v = 0; v < 8; ++v)
        sqr[v] = sq[b * S_SEQ + rBase + v + half * 8];

    const bf16_t* xa = xbf + (xrow0 + ln) * E_DIM;     // A: lane = query row

    for (int c2 = 0; c2 < 4; ++c2) {                   // 8 key-tiles per chunk
        v8f acc[8];
#pragma unroll
        for (int j = 0; j < 8; ++j) acc[j] = vzero;
        for (int k = 0; k < E_DIM; k += 32) {
            v16bf a = load16(xa + k + half * 8, xa + k + 16 + half * 8);
#pragma unroll
            for (int j = 0; j < 8; ++j) {
                const int kt = (c2 * 8 + j) * 2 + ch;  // interleaved key tiles
                const bf16_t* xb = xbf +
                    (size_t)(b * S_SEQ + kt * 16 + ln) * E_DIM + k + half * 8;
                v16bf bf = load16(xb, xb + 16);
                acc[j] = __builtin_amdgcn_wmma_f32_16x16x32_bf16(
                    false, a, false, bf, (short)0, acc[j], false, false);
            }
        }
#pragma unroll
        for (int j = 0; j < 8; ++j) {
            const int   kt  = (c2 * 8 + j) * 2 + ch;
            const float sqk = sq[b * S_SEQ + kt * 16 + ln];
#pragma unroll
            for (int v = 0; v < 8; ++v) {
                const int row = rt * 16 + v + half * 8;
                logits[row * LSTRIDE + kt * 16 + ln] =
                    (2.0f * acc[j][v] - sqr[v] - sqk) * TEMP_INV;
            }
        }
    }
    __syncthreads();

    // ===== Phase 2: softmax numerator in place; 1/sum stashed per row =======
    {
        const int row = tid >> 2;          // 4 threads per row, 64 rows
        const int seg = tid & 3;
        float* lrow = logits + row * LSTRIDE + seg * 256;
        float m = -3.4e38f;
        for (int c = 0; c < 256; ++c) m = fmaxf(m, lrow[c]);
        m = fmaxf(m, __shfl_xor(m, 1, 32));
        m = fmaxf(m, __shfl_xor(m, 2, 32));
        float s = 0.0f;
        for (int c = 0; c < 256; ++c) {
            const float e = __expf(lrow[c] - m);
            lrow[c] = e;                   // in-place: same thread, same cells
            s += e;
        }
        s += __shfl_xor(s, 1, 32);
        s += __shfl_xor(s, 2, 32);
        if (seg == 0) rowsum[row] = 1.0f / s;
    }
    __syncthreads();

    // ====== Phase 3: h = exp(l) @ w1^T, scaled by 1/sum at store (+b1) ======
    {
        v8f hacc[16];
#pragma unroll
        for (int ct = 0; ct < 16; ++ct) hacc[ct] = vzero;
        const float* pa = logits + (rt * 16 + ln) * LSTRIDE;  // A from LDS f32
        for (int k = 0; k < S_SEQ; k += 32) {
            v16bf a = cvt16(pa + k + half * 8, pa + k + 16 + half * 8);
#pragma unroll
            for (int ct = 0; ct < 16; ++ct) {
                const int col = ch * 256 + ct * 16 + ln;      // lane = h column
                const bf16_t* wb = w1bf + (size_t)col * S_SEQ + k + half * 8;
                v16bf bf = load16(wb, wb + 16);
                hacc[ct] = __builtin_amdgcn_wmma_f32_16x16x32_bf16(
                    false, a, false, bf, (short)0, hacc[ct], false, false);
            }
        }
        // per-row softmax denominators (rowsum is outside the aliased region)
        float rinv[8];
#pragma unroll
        for (int v = 0; v < 8; ++v) rinv[v] = rowsum[rt * 16 + v + half * 8];

        __syncthreads();   // all p reads done -> logits region is dead
#pragma unroll
        for (int ct = 0; ct < 16; ++ct) {
            const int   col  = ch * 256 + ct * 16 + ln;
            const float bias = b1[col];
#pragma unroll
            for (int v = 0; v < 8; ++v) {
                const int row = rt * 16 + v + half * 8;
                hbuf[row * HSTRIDE + col] = hacc[ct][v] * rinv[v] + bias;
            }
        }
    }
    __syncthreads();

    // ================= Phase 4: LayerNorm -> relu -> fc2 -> relu ============
    {
        const int row = tid >> 2;          // 64 rows, 4 threads each
        const int seg = tid & 3;
        const float* hrow = hbuf + row * HSTRIDE + seg * 128;
        float s0 = 0.f, s1 = 0.f;
        for (int c = 0; c < 128; ++c) {
            const float v = hrow[c];
            s0 += v;
            s1 += v * v;
        }
        s0 += __shfl_xor(s0, 1, 32); s0 += __shfl_xor(s0, 2, 32);
        s1 += __shfl_xor(s1, 1, 32); s1 += __shfl_xor(s1, 2, 32);
        const float mu  = s0 * (1.0f / H_DIM);
        const float var = s1 * (1.0f / H_DIM) - mu * mu;
        const float rs  = rsqrtf(var + 1e-5f);
        float acc2 = 0.f;
        const int cb = seg * 128;
        for (int c = 0; c < 128; ++c) {
            const float lnv = (hrow[c] - mu) * rs * g1[cb + c] + beta1[cb + c];
            acc2 += fmaxf(lnv, 0.0f) * w2[cb + c];
        }
        acc2 += __shfl_xor(acc2, 1, 32);
        acc2 += __shfl_xor(acc2, 2, 32);
        if (seg == 0) {
            const float o = acc2 + b2[0];
            out[(size_t)b * S_SEQ + rb * BM + row] = fmaxf(o, 0.0f);
        }
    }
}

extern "C" void kernel_launch(void* const* d_in, const int* in_sizes, int n_in,
                              void* d_out, int out_size, void* d_ws, size_t ws_size,
                              hipStream_t stream) {
    (void)in_sizes; (void)n_in; (void)out_size; (void)ws_size;
    const float* x     = (const float*)d_in[0];
    const float* w1    = (const float*)d_in[1];
    const float* b1    = (const float*)d_in[2];
    const float* g1    = (const float*)d_in[3];
    const float* beta1 = (const float*)d_in[4];
    const float* w2    = (const float*)d_in[5];
    const float* b2    = (const float*)d_in[6];
    float* out = (float*)d_out;

    const size_t xElems  = (size_t)N_B * S_SEQ * E_DIM;   // 33,554,432
    const size_t w1Elems = (size_t)H_DIM * S_SEQ;         // 524,288
    bf16_t* xbf  = (bf16_t*)d_ws;                                          // 64 MB
    bf16_t* w1bf = (bf16_t*)((char*)d_ws + xElems * sizeof(bf16_t));       // +1 MB
    float*  sq   = (float*)((char*)d_ws + (xElems + w1Elems) * sizeof(bf16_t));

    conv_x_kernel<<<N_B * S_SEQ, 128, 0, stream>>>(x, xbf, sq);
    conv_w1_kernel<<<(int)((w1Elems + 255) / 256), 256, 0, stream>>>(w1, w1bf, (int)w1Elems);
    fused_kernel<<<N_B * (S_SEQ / BM), 256, SMEM_BYTES, stream>>>(
        xbf, sq, w1bf, b1, g1, beta1, w2, b2, out);
}
